// CausalSelfAttention_72722386256234
// MI455X (gfx1250) — matrix operand order repaired and checked by reading, not compiled, not run on hardware
//
#include <hip/hip_runtime.h>
#include <hip/hip_bf16.h>

typedef __bf16 bf16;
typedef __attribute__((ext_vector_type(16))) __bf16 bf16x16;
typedef __attribute__((ext_vector_type(8)))  __bf16 bf16x8;
typedef __attribute__((ext_vector_type(8)))  float   f32x8;

union BF16x16 { bf16x16 v16; bf16x8 v8[2]; };

#define WMMA_BF16(A_, B_, C_) \
  __builtin_amdgcn_wmma_f32_16x16x32_bf16(false, (A_), false, (B_), (short)0, (C_), false, false)

__device__ inline f32x8 zero8() {
  f32x8 z = {0.f, 0.f, 0.f, 0.f, 0.f, 0.f, 0.f, 0.f};
  return z;
}

// ---- CDNA5 async memory->LDS path (ASYNCcnt-tracked, no VGPR round-trip) ----
__device__ inline void async_copy16(const bf16* gsrc, bf16* ldst) {
  // AS3->flat cast puts the LDS byte offset in the low 32 bits of the flat addr.
  unsigned lds = (unsigned)(uintptr_t)ldst;
  asm volatile("global_load_async_to_lds_b128 %0, %1, off"
               :: "v"(lds), "v"(gsrc) : "memory");
}
__device__ inline void wait_async0() { asm volatile("s_wait_asynccnt 0x0" ::: "memory"); }
__device__ inline void wait_ds0()    { asm volatile("s_wait_dscnt 0x0"    ::: "memory"); }

// ---------------------------------------------------------------------------
// fp32 -> bf16 conversion (vector x4)
// ---------------------------------------------------------------------------
__global__ void __launch_bounds__(256) cvt_f32_to_bf16(const float* __restrict__ in,
                                                       bf16* __restrict__ out, int n4) {
  int i = blockIdx.x * blockDim.x + threadIdx.x;
  if (i < n4) {
    float4 v = *(const float4*)(in + (size_t)i * 4);
    bf16* o = out + (size_t)i * 4;
    o[0] = (bf16)v.x; o[1] = (bf16)v.y; o[2] = (bf16)v.z; o[3] = (bf16)v.w;
  }
}

// ---------------------------------------------------------------------------
// 128x128 block GEMM core: C[m,n] = sum_k A[m,k] * B[n,k]   (both K-major, bf16)
// 256 threads = 8 waves; wave (wm,wn) in 2x4 grid computes 64(M) x 32(N).
// Double-buffered LDS; tile k+1 staged via async-to-LDS while computing tile k.
// K loop fully unrolled (K is a compile-time literal at both call sites).
// LDS rows padded to 40 elems (80B): 16B-aligned rows, conflict-free strides.
// ---------------------------------------------------------------------------
#define LDT 40
#define TILE_ELEMS (128 * LDT)

__device__ inline void stage_tile_async(const bf16* __restrict__ A,
                                        const bf16* __restrict__ B,
                                        int K, int aRow0, int bRow0, int k0,
                                        bf16* ldsA, bf16* ldsB) {
  const int tid = threadIdx.x;
#pragma unroll
  for (int i = 0; i < 2; ++i) {
    int c   = tid * 2 + i;        // 0..511
    int row = c >> 2;             // 0..127
    int kc  = (c & 3) * 8;        // 0,8,16,24
    async_copy16(A + (size_t)(aRow0 + row) * K + k0 + kc, ldsA + row * LDT + kc);
    async_copy16(B + (size_t)(bRow0 + row) * K + k0 + kc, ldsB + row * LDT + kc);
  }
}

__device__ __forceinline__ void gemm128x128(const bf16* __restrict__ A,
                                            const bf16* __restrict__ B,
                                            const int K, int aRow0, int bRow0,
                                            bf16* ldsA, bf16* ldsB, f32x8 acc[4][2]) {
  const int tid  = threadIdx.x;
  const int wave = tid >> 5;
  const int lane = tid & 31;
  const int wm = wave >> 2;   // 0..1
  const int wn = wave & 3;    // 0..3
  const int h  = lane >> 4;   // lane half
  const int r  = lane & 15;

#pragma unroll
  for (int ms = 0; ms < 4; ++ms)
#pragma unroll
    for (int ns = 0; ns < 2; ++ns)
      acc[ms][ns] = zero8();

  stage_tile_async(A, B, K, aRow0, bRow0, 0, ldsA, ldsB);
  wait_async0();
  __syncthreads();

#pragma unroll
  for (int k0 = 0; k0 < K; k0 += 32) {
    const int cur = (k0 >> 5) & 1;
    bf16* curA = ldsA + cur * TILE_ELEMS;
    bf16* curB = ldsB + cur * TILE_ELEMS;
    if (k0 + 32 < K)   // statically resolved per unrolled iteration
      stage_tile_async(A, B, K, aRow0, bRow0, k0 + 32,
                       ldsA + (cur ^ 1) * TILE_ELEMS, ldsB + (cur ^ 1) * TILE_ELEMS);

    BF16x16 a[4], b[2];
#pragma unroll
    for (int ms = 0; ms < 4; ++ms) {
      const bf16* p = curA + (wm * 64 + ms * 16 + r) * LDT + h * 8;
      a[ms].v8[0] = *(const bf16x8*)(p);        // K = h*8 .. h*8+7
      a[ms].v8[1] = *(const bf16x8*)(p + 16);   // K = 16+h*8 .. 16+h*8+7
    }
#pragma unroll
    for (int ns = 0; ns < 2; ++ns) {
      const bf16* p = curB + (wn * 32 + ns * 16 + r) * LDT + h * 8;
      b[ns].v8[0] = *(const bf16x8*)(p);        // K = h*8 .. h*8+7
      b[ns].v8[1] = *(const bf16x8*)(p + 16);   // K = 16+h*8 .. 16+h*8+7
    }
#pragma unroll
    for (int ms = 0; ms < 4; ++ms)
#pragma unroll
      for (int ns = 0; ns < 2; ++ns)
        acc[ms][ns] = WMMA_BF16(a[ms].v16, b[ns].v16, acc[ms][ns]);

    wait_ds0();      // my LDS reads landed in VGPRs (WAR vs next stage)
    wait_async0();   // my async stores into the other buffer completed
    __syncthreads(); // everyone done -> safe to compute next / restage
  }
}

// ---------------------------------------------------------------------------
// QKV projection: [4096,1024] x [3072,1024]^T -> Q/K [B,H,T,D], V^T [B,H,D,T]
// Q pre-scaled by 1/sqrt(64).
// ---------------------------------------------------------------------------
__global__ void __launch_bounds__(256) qkv_gemm_kernel(const bf16* __restrict__ Xb,
                                                       const bf16* __restrict__ Wb,
                                                       bf16* __restrict__ Q,
                                                       bf16* __restrict__ Kt,
                                                       bf16* __restrict__ Vt) {
  __shared__ bf16 ldsA[2 * TILE_ELEMS];
  __shared__ bf16 ldsB[2 * TILE_ELEMS];
  f32x8 acc[4][2];
  gemm128x128(Xb, Wb, 1024, blockIdx.y * 128, blockIdx.x * 128, ldsA, ldsB, acc);

  const int lane = threadIdx.x & 31, wave = threadIdx.x >> 5;
  const int wm = wave >> 2, wn = wave & 3;
  const int h = lane >> 4, cn = lane & 15;

  for (int ms = 0; ms < 4; ++ms)
    for (int ns = 0; ns < 2; ++ns)
      for (int j = 0; j < 8; ++j) {
        int m = blockIdx.y * 128 + wm * 64 + ms * 16 + h * 8 + j;
        int n = blockIdx.x * 128 + wn * 32 + ns * 16 + cn;
        float v = acc[ms][ns][j];
        int b = m >> 11, t = m & 2047;
        int which = n >> 10;          // 0=Q 1=K 2=V
        int hd = n & 1023;
        int hh = hd >> 6, d = hd & 63;
        size_t bh = (size_t)(b * 16 + hh);
        if (which == 0)      Q [(bh * 2048 + t) * 64 + d] = (bf16)(v * 0.125f);
        else if (which == 1) Kt[(bh * 2048 + t) * 64 + d] = (bf16)v;
        else                 Vt[(bh * 64 + d) * 2048 + t] = (bf16)v;
      }
}

// ---------------------------------------------------------------------------
// Flash attention: block = (bh, qTile of 128). 8 waves x 16 queries each.
// K/V tiles staged via async-to-LDS.
// ---------------------------------------------------------------------------
__global__ void __launch_bounds__(256) attn_kernel(const bf16* __restrict__ Q,
                                                   const bf16* __restrict__ Kt,
                                                   const bf16* __restrict__ Vt,
                                                   bf16* __restrict__ O) {
  const int bh = blockIdx.x;     // 0..31
  const int qt = blockIdx.y;     // 0..15
  __shared__ bf16 kTile[32 * 72];       // 32 keys x 64 d, stride 72
  __shared__ bf16 vTile[64 * LDT];      // 64 d x 32 keys, stride 40
  __shared__ bf16 pTile[8][16 * LDT];   // per-wave P scratch

  const int tid = threadIdx.x, wave = tid >> 5, lane = tid & 31;
  const int h = lane >> 4, r = lane & 15;
  const int qRow0 = qt * 128 + wave * 16;

  // Q A-operands for this wave's 16 rows (d split 0..31 / 32..63), prescaled.
  BF16x16 qa[2];
  {
    const bf16* qp = Q + ((size_t)bh * 2048 + qRow0 + r) * 64;
    qa[0].v8[0] = *(const bf16x8*)(qp + h * 8);
    qa[0].v8[1] = *(const bf16x8*)(qp + 16 + h * 8);
    qa[1].v8[0] = *(const bf16x8*)(qp + 32 + h * 8);
    qa[1].v8[1] = *(const bf16x8*)(qp + 48 + h * 8);
  }

  f32x8 o[4];
#pragma unroll
  for (int ns = 0; ns < 4; ++ns) o[ns] = zero8();
  float ms_[8], ls_[8];
#pragma unroll
  for (int j = 0; j < 8; ++j) { ms_[j] = -1e30f; ls_[j] = 0.f; }

  const int nkt = (qt + 1) * 4;   // key tiles of 32, causal upper bound for block
  for (int kt = 0; kt < nkt; ++kt) {
    const int kBase = kt * 32;
    wait_ds0();          // prior LDS reads complete before restaging
    __syncthreads();
    {   // async stage K (32x64) and V^T (64x32)
      int key = tid >> 3, dc = (tid & 7) * 8;
      async_copy16(Kt + ((size_t)bh * 2048 + kBase + key) * 64 + dc,
                   &kTile[key * 72 + dc]);
      int d = tid >> 2, kc = (tid & 3) * 8;
      async_copy16(Vt + ((size_t)bh * 64 + d) * 2048 + kBase + kc,
                   &vTile[d * LDT + kc]);
    }
    wait_async0();
    __syncthreads();

    if (kBase <= qRow0 + 15) {          // wave-uniform causal skip
      bf16* pw = &pTile[wave][0];
#pragma unroll
      for (int g = 0; g < 2; ++g) {
        const int keyG = kBase + g * 16;
        if (keyG <= qRow0 + 15) {
          f32x8 s = zero8();
          BF16x16 kb;
          const bf16* kp = &kTile[(g * 16 + r) * 72];
          kb.v8[0] = *(const bf16x8*)(kp + h * 8);
          kb.v8[1] = *(const bf16x8*)(kp + 16 + h * 8);
          s = WMMA_BF16(qa[0].v16, kb.v16, s);
          kb.v8[0] = *(const bf16x8*)(kp + 32 + h * 8);
          kb.v8[1] = *(const bf16x8*)(kp + 48 + h * 8);
          s = WMMA_BF16(qa[1].v16, kb.v16, s);

          const int keyCol = keyG + r;  // this lane's key column
#pragma unroll
          for (int j = 0; j < 8; ++j) {
            int qrow = qRow0 + h * 8 + j;
            float sv = (keyCol <= qrow) ? s[j] : -1e30f;
            float mx = sv;
            for (int off = 1; off < 16; off <<= 1)
              mx = fmaxf(mx, __shfl_xor(mx, off, 32));
            float mNew = fmaxf(ms_[j], mx);
            float p = __expf(sv - mNew);
            float sum = p;
            for (int off = 1; off < 16; off <<= 1)
              sum += __shfl_xor(sum, off, 32);
            float corr = __expf(ms_[j] - mNew);
            ls_[j] = ls_[j] * corr + sum;
            ms_[j] = mNew;
            for (int ns = 0; ns < 4; ++ns) o[ns][j] *= corr;
            pw[(h * 8 + j) * LDT + g * 16 + r] = (bf16)p;
          }
        } else {
          for (int j = 0; j < 8; ++j)
            pw[(h * 8 + j) * LDT + g * 16 + r] = (bf16)0.f;
        }
      }
      // O += P(16x32) x V(32x64)
      BF16x16 pa;
      const bf16* pp = &pTile[wave][0] + r * LDT;
      pa.v8[0] = *(const bf16x8*)(pp + h * 8);
      pa.v8[1] = *(const bf16x8*)(pp + 16 + h * 8);
#pragma unroll
      for (int ns = 0; ns < 4; ++ns) {
        BF16x16 vb;
        const bf16* vp = &vTile[(ns * 16 + r) * LDT + h * 8];
        vb.v8[0] = *(const bf16x8*)(vp);
        vb.v8[1] = *(const bf16x8*)(vp + 16);
        o[ns] = WMMA_BF16(pa.v16, vb.v16, o[ns]);
      }
    }
  }

  // Normalize and store O as [B,T,H,D] bf16 (== [B,T,C] rows for out-proj GEMM)
  const int b = bh >> 4, hh = bh & 15;
  for (int j = 0; j < 8; ++j) {
    float inv = 1.0f / ls_[j];
    int qrow = qRow0 + h * 8 + j;
    for (int ns = 0; ns < 4; ++ns) {
      int d = ns * 16 + r;
      O[(((size_t)b * 2048 + qrow) * 16 + hh) * 64 + d] = (bf16)(o[ns][j] * inv);
    }
  }
}

// ---------------------------------------------------------------------------
// Output projection: [4096,1024] x [1024,1024]^T -> fp32 d_out
// ---------------------------------------------------------------------------
__global__ void __launch_bounds__(256) out_gemm_kernel(const bf16* __restrict__ Ob,
                                                       const bf16* __restrict__ Wb,
                                                       float* __restrict__ out) {
  __shared__ bf16 ldsA[2 * TILE_ELEMS];
  __shared__ bf16 ldsB[2 * TILE_ELEMS];
  f32x8 acc[4][2];
  gemm128x128(Ob, Wb, 1024, blockIdx.y * 128, blockIdx.x * 128, ldsA, ldsB, acc);

  const int lane = threadIdx.x & 31, wave = threadIdx.x >> 5;
  const int wm = wave >> 2, wn = wave & 3;
  const int h = lane >> 4, cn = lane & 15;
  for (int ms = 0; ms < 4; ++ms)
    for (int ns = 0; ns < 2; ++ns)
      for (int j = 0; j < 8; ++j) {
        int m = blockIdx.y * 128 + wm * 64 + ms * 16 + h * 8 + j;
        int n = blockIdx.x * 128 + wn * 32 + ns * 16 + cn;
        out[(size_t)m * 1024 + n] = acc[ms][ns][j];
      }
}

// ---------------------------------------------------------------------------
extern "C" void kernel_launch(void* const* d_in, const int* in_sizes, int n_in,
                              void* d_out, int out_size, void* d_ws, size_t ws_size,
                              hipStream_t stream) {
  const float* x     = (const float*)d_in[0];   // [2,2048,1024]
  const float* w_qkv = (const float*)d_in[1];   // [3072,1024]
  const float* w_out = (const float*)d_in[2];   // [1024,1024]
  float* out = (float*)d_out;                   // [2,2048,1024]

  const size_t NX = 4194304, NWQ = 3145728, NWO = 1048576;
  char* w = (char*)d_ws;
  bf16* Xb    = (bf16*)w; w += NX  * 2;
  bf16* Wqkvb = (bf16*)w; w += NWQ * 2;
  bf16* Woutb = (bf16*)w; w += NWO * 2;
  bf16* Qb    = (bf16*)w; w += NX * 2;   // [B,H,T,D]
  bf16* Kb    = (bf16*)w; w += NX * 2;   // [B,H,T,D]
  bf16* Vtb   = (bf16*)w; w += NX * 2;   // [B,H,D,T]
  bf16* Ob    = (bf16*)w; w += NX * 2;   // [B,T,H,D]

  cvt_f32_to_bf16<<<(int)(NX  / 4 + 255) / 256, 256, 0, stream>>>(x,     Xb,    (int)(NX  / 4));
  cvt_f32_to_bf16<<<(int)(NWQ / 4 + 255) / 256, 256, 0, stream>>>(w_qkv, Wqkvb, (int)(NWQ / 4));
  cvt_f32_to_bf16<<<(int)(NWO / 4 + 255) / 256, 256, 0, stream>>>(w_out, Woutb, (int)(NWO / 4));

  qkv_gemm_kernel<<<dim3(24, 32), 256, 0, stream>>>(Xb, Wqkvb, Qb, Kb, Vtb);
  attn_kernel    <<<dim3(32, 16), 256, 0, stream>>>(Qb, Kb, Vtb, Ob);
  out_gemm_kernel<<<dim3(8, 32),  256, 0, stream>>>(Ob, Woutb, out);
}
